// GCPN_CReM_50775103373332
// MI455X (gfx1250) — matrix operand order repaired and checked by reading, not compiled, not run on hardware
//
#include <hip/hip_runtime.h>
#include <hip/hip_bf16.h>
#include <stdint.h>

// ---------------------------------------------------------------------------
// Problem constants
// ---------------------------------------------------------------------------
#define B_GRAPHS   2048
#define N_CAND     524288
#define EMB        128
#define KDIM       256          // 2*EMB == HID == 256 (uniform GEMM K/N)
#define NDIM       256
#define TILE_M     256          // rows per workgroup (2 strips per wave)
#define WAVE_M     16           // rows per strip
#define STRIPS     2            // strips per wave (8 waves * 2 * 16 = 256)
#define KT_STEPS   8            // 256 / 32
#define NT_TILES   16           // 256 / 16
#define WPACK_ELEMS 65536       // 256*256 per layer

typedef __attribute__((ext_vector_type(16))) __bf16 v16bf;
typedef __attribute__((ext_vector_type(8)))  float  v8f;

// ---------------------------------------------------------------------------
// Kernel 0: weight pack (fp32 -> bf16, WMMA B-fragment order), g_emb copy,
//           denominator zeroing.
// Pack layout (per layer): frag f = kt*16+nt, per lane l 16 contiguous bf16:
//   value[e] = W[kt*32 + (l>=16?8:0) + (e<8 ? e : e+8)][nt*16 + (l&15)]
// flat index = ((f*32 + l)*16 + e)
// ---------------------------------------------------------------------------
__global__ void prep_kernel(const float* __restrict__ g_emb,
                            const float* __restrict__ W1,
                            const float* __restrict__ W2,
                            const float* __restrict__ W3,
                            __bf16* __restrict__ wpack,
                            float* __restrict__ g_out,
                            float* __restrict__ denom)
{
    const int tid = blockIdx.x * blockDim.x + threadIdx.x;

    if (tid < B_GRAPHS) denom[tid] = 0.0f;

    if (tid < (B_GRAPHS * EMB / 4)) {
        ((float4*)g_out)[tid] = ((const float4*)g_emb)[tid];
    }

    if (tid < 3 * WPACK_ELEMS) {
        const int layer = tid >> 16;          // /65536
        const int r     = tid & 65535;
        const int f     = r >> 9;             // /512
        const int le    = r & 511;
        const int l     = le >> 4;
        const int e     = le & 15;
        const int kt    = f >> 4;
        const int nt    = f & 15;
        const int k     = kt * 32 + ((l >= 16) ? 8 : 0) + ((e < 8) ? e : e + 8);
        const int n     = nt * 16 + (l & 15);
        const float* W  = (layer == 0) ? W1 : (layer == 1) ? W2 : W3;
        wpack[tid] = (__bf16)W[k * NDIM + n];
    }
}

// ---------------------------------------------------------------------------
// Kernel 1: fused gather-concat + 3x (GEMM 256x256 + bias + ReLU) + final dot
//           + exp + segment-sum atomics.  All activations stay in LDS.
// ---------------------------------------------------------------------------
__global__ __launch_bounds__(256, 1)
void mlp_fused(const float* __restrict__ g_emb,
               const float* __restrict__ cand_emb,
               const float* __restrict__ b1,
               const float* __restrict__ b2,
               const float* __restrict__ b3,
               const float* __restrict__ Wf,
               const float* __restrict__ bf_p,
               const int*   __restrict__ batch_idx,
               const __bf16* __restrict__ wpack,
               float* __restrict__ Xout,
               float* __restrict__ e_ws,
               float* __restrict__ denom)
{
    __shared__ __bf16 act[TILE_M * KDIM];     // 128 KB activations (bf16, row-major)
    __shared__ __bf16 wbuf[KDIM * NDIM];      // 128 KB packed weight fragments
    __shared__ float  sbias[NDIM];            // 1 KB bias / final weight

    const int t      = threadIdx.x;
    const int block0 = blockIdx.x * TILE_M;

    // ---- Stage 0: build X tile -> Xout (fp32) + act (bf16) -----------------
    #pragma unroll 1
    for (int rr = 0; rr < 2; ++rr) {
        const int row  = rr * 128 + (t >> 1);    // 0..255
        const int half = t & 1;                  // 0 = graph emb, 1 = candidate
        const int grow = block0 + row;
        const int bidx = batch_idx[grow];
        const float4* src4 = (const float4*)(half
                              ? (cand_emb + (size_t)grow * EMB)
                              : (g_emb    + (size_t)bidx * EMB));
        float4* dst4  = (float4*)(Xout + (size_t)grow * KDIM + half * EMB);
        uint4*  lds16 = (uint4*)&act[row * KDIM + half * EMB];
        #pragma unroll
        for (int i = 0; i < 16; ++i) {
            float4 a = src4[2 * i];
            float4 b = src4[2 * i + 1];
            dst4[2 * i]     = a;
            dst4[2 * i + 1] = b;
            union { __bf16 h[8]; uint4 q; } pk;
            pk.h[0] = (__bf16)a.x; pk.h[1] = (__bf16)a.y;
            pk.h[2] = (__bf16)a.z; pk.h[3] = (__bf16)a.w;
            pk.h[4] = (__bf16)b.x; pk.h[5] = (__bf16)b.y;
            pk.h[6] = (__bf16)b.z; pk.h[7] = (__bf16)b.w;
            lds16[i] = pk.q;
        }
    }
    __syncthreads();

    const int wave  = t >> 5;          // 0..7
    const int lane  = t & 31;
    const int lhalf = lane >> 4;       // lane group 0/1
    const int l15   = lane & 15;

    // ---- 3 fused GEMM + ReLU layers ---------------------------------------
    #pragma unroll 1
    for (int layer = 0; layer < 3; ++layer) {
        // stage packed weights (128 KB) + bias into LDS, coalesced
        const uint4* wsrc = (const uint4*)(wpack + (size_t)layer * WPACK_ELEMS);
        uint4* wdst = (uint4*)wbuf;
        #pragma unroll
        for (int i = 0; i < 32; ++i)
            wdst[i * 256 + t] = wsrc[i * 256 + t];
        const float* bp = (layer == 0) ? b1 : (layer == 1) ? b2 : b3;
        sbias[t] = bp[t];
        __syncthreads();

        // each wave computes two 16-row strips against all 256 columns
        #pragma unroll 1
        for (int s = 0; s < STRIPS; ++s) {
            const int m0 = (wave * STRIPS + s) * WAVE_M;

            // accumulators: 16 N-tiles x 8 VGPRs, seeded with bias
            v8f acc[NT_TILES];
            #pragma unroll
            for (int nt = 0; nt < NT_TILES; ++nt) {
                const float bv = sbias[nt * 16 + l15];
                #pragma unroll
                for (int r = 0; r < 8; ++r) acc[nt][r] = bv;
            }

            // K loop kept rolled: bounds VGPR pressure (one K-step of B
            // fragments live at a time -> no >256-VGPR MSB addressing)
            #pragma unroll 1
            for (int kt = 0; kt < KT_STEPS; ++kt) {
                // A fragment: 16x32 bf16 per ISA layout.
                // lane (l<16):  row m0+l,      K = kt*32 + {0..7, 16..23}
                // lane (l>=16): row m0+(l-16), K = kt*32 + {8..15, 24..31}
                union { v16bf v; uint4 q[2]; } af;
                const int arow  = m0 + l15;
                const int kbase = kt * 32 + lhalf * 8;
                af.q[0] = *(const uint4*)&act[arow * KDIM + kbase];
                af.q[1] = *(const uint4*)&act[arow * KDIM + kbase + 16];

                #pragma unroll
                for (int nt = 0; nt < NT_TILES; ++nt) {
                    union { v16bf v; uint4 q[2]; } bfm;
                    const uint4* bpq =
                        (const uint4*)&wbuf[(((kt * NT_TILES + nt) * 32) + lane) * 16];
                    bfm.q[0] = bpq[0];
                    bfm.q[1] = bpq[1];
                    acc[nt] = __builtin_amdgcn_wmma_f32_16x16x32_bf16(
                        /*neg_a=*/false, af.v, /*neg_b=*/false, bfm.v,
                        /*c_mod=*/(short)0, acc[nt],
                        /*reuse_a=*/false, /*reuse_b=*/false);
                }
            }

            // ReLU + write back this wave's strip (strip-local: no race)
            #pragma unroll
            for (int nt = 0; nt < NT_TILES; ++nt) {
                const int col = nt * 16 + l15;
                #pragma unroll
                for (int r = 0; r < 8; ++r) {
                    float v = acc[nt][r];
                    v = v > 0.0f ? v : 0.0f;
                    const int rowr = m0 + lhalf * 8 + r;    // C/D layout rows
                    act[rowr * KDIM + col] = (__bf16)v;
                }
            }
        }
        __syncthreads();   // everyone done with wbuf/sbias before next layer
    }

    // ---- Final: logits = h3 . Wf + bf, e = exp, segment-sum atomics --------
    sbias[t] = Wf[t];
    __syncthreads();

    #pragma unroll 1
    for (int s = 0; s < STRIPS; ++s) {
        const int row = (wave * STRIPS + s) * WAVE_M + l15;    // own strip
        const int k0  = lhalf * 128;                           // lane pair splits K
        float part = 0.0f;
        #pragma unroll 8
        for (int k = 0; k < 128; ++k)
            part += (float)act[row * KDIM + k0 + k] * sbias[k0 + k];
        part += __shfl_xor(part, 16, 32);          // combine K halves (wave32)
        if (lhalf == 0) {
            const int grow  = block0 + row;
            const float lg  = part + bf_p[0];
            const float ev  = __expf(lg);
            e_ws[grow] = ev;
            atomicAdd(&denom[batch_idx[grow]], ev);
        }
    }
}

// ---------------------------------------------------------------------------
// Kernel 2: probs = e / denom[batch_idx]
// ---------------------------------------------------------------------------
__global__ void probs_kernel(const float* __restrict__ e_ws,
                             const float* __restrict__ denom,
                             const int*   __restrict__ batch_idx,
                             float* __restrict__ probs)
{
    const int i = blockIdx.x * blockDim.x + threadIdx.x;
    if (i < N_CAND) probs[i] = e_ws[i] / denom[batch_idx[i]];
}

// ---------------------------------------------------------------------------
// Launcher
// ---------------------------------------------------------------------------
extern "C" void kernel_launch(void* const* d_in, const int* in_sizes, int n_in,
                              void* d_out, int out_size, void* d_ws, size_t ws_size,
                              hipStream_t stream)
{
    const float* g_emb    = (const float*)d_in[0];
    const float* cand_emb = (const float*)d_in[1];
    const float* W1       = (const float*)d_in[2];
    const float* b1       = (const float*)d_in[3];
    const float* W2       = (const float*)d_in[4];
    const float* b2       = (const float*)d_in[5];
    const float* W3       = (const float*)d_in[6];
    const float* b3       = (const float*)d_in[7];
    const float* Wf       = (const float*)d_in[8];
    const float* bf       = (const float*)d_in[9];
    const int*   bidx     = (const int*)d_in[10];

    float* out   = (float*)d_out;
    float* g_out = out;                                      // [2048*128]
    float* X_out = out + (size_t)B_GRAPHS * EMB;             // [N*256]
    float* p_out = X_out + (size_t)N_CAND * KDIM;            // [N]

    __bf16* wpack = (__bf16*)d_ws;                           // 3*65536 bf16
    float*  e_ws  = (float*)((char*)d_ws + (size_t)3 * WPACK_ELEMS * 2);
    float*  denom = e_ws + N_CAND;                           // [2048]

    prep_kernel<<<768, 256, 0, stream>>>(g_emb, W1, W2, W3, wpack, g_out, denom);
    mlp_fused<<<N_CAND / TILE_M, 256, 0, stream>>>(
        g_emb, cand_emb, b1, b2, b3, Wf, bf, bidx, wpack, X_out, e_ws, denom);
    probs_kernel<<<N_CAND / 256, 256, 0, stream>>>(e_ws, denom, bidx, p_out);
}
